// Concept_model_171798691895
// MI455X (gfx1250) — matrix operand order repaired and checked by reading, not compiled, not run on hardware
//
#include <hip/hip_runtime.h>
#include <hip/hip_bf16.h>

typedef __attribute__((ext_vector_type(16))) _Float16 v16h;
typedef __attribute__((ext_vector_type(8)))  float    v8f;

#define BATCH 16384
#define TLEN  64

// Packed-f16 embedding table layout in d_ws (dword offsets).
// All uint2 (b64) entry arrays are 8-byte aligned.
#define OFF12D 0    // 4  x uint   : (e1,e2) pair, idx = i0*2+i1
#define OFF56D 4    // 10 x uint   : (e5,e6) pair, idx = i4*2+i5
#define OFF3D  16   // 21 x uint2  : e3 triple padded to 4xf16
#define OFF4D  60   // 22 x uint2  : e4 triple
#define OFF7D  104  // 22 x uint2  : e7 triple
#define OFF8D  148  // 24 x uint2  : e8 triple  (total 196 dwords = 784 B)

// gfx1250 hardware v_tanh_f32 (confirmed in round-2 disasm).
#if __has_builtin(__builtin_amdgcn_tanhf)
__device__ __forceinline__ float fast_tanh(float xv) {
    return __builtin_amdgcn_tanhf(xv);
}
#else
__device__ __forceinline__ float fast_tanh(float xv) {
    return 2.f * __builtin_amdgcn_rcpf(1.f + __expf(-2.f * xv)) - 1.f;
}
#endif
__device__ __forceinline__ float fast_sig(float xv) {
    return 0.5f + 0.5f * fast_tanh(0.5f * xv);
}

__device__ __forceinline__ unsigned pk2(float a, float b) {
    union { _Float16 h[2]; unsigned u; } r;
    r.h[0] = (_Float16)a; r.h[1] = (_Float16)b;
    return r.u;
}
__device__ __forceinline__ float unpk0(unsigned u) {
    union { unsigned u; _Float16 h[2]; } r; r.u = u; return (float)r.h[0];
}
__device__ __forceinline__ float unpk1(unsigned u) {
    union { unsigned u; _Float16 h[2]; } r; r.u = u; return (float)r.h[1];
}

// Permuted A-row -> gate index. Rows [0..15] = [i0..i3,f0..f3,i4..i7,f4..f7]
// (g|o matrix: +16). Lane n's 8 D-registers then hold complete i,f (resp. g,o)
// for hidden units 0..3 (low half) / 4..7 (high half) of batch column n:
// the cell update needs NO cross-lane traffic.
__device__ __forceinline__ int gate_of(int mm) {
    return ((mm >> 2) & 1) * 8 + ((mm >> 3) & 1) * 4 + (mm & 3);
}

union BOp { v16h h; unsigned u[8]; };

struct EmbP { unsigned u0, u1, u2, u3, u4, u5, u6, u7; };

// 8 loads: 2x b128 (indices) + 2x b32 (fused pair tables) + 4x b64 (triples).
__device__ __forceinline__ EmbP load_emb(const int* __restrict__ x, int bidx, int t,
                                         const unsigned* __restrict__ ws) {
    const int4* xp = (const int4*)(x + (bidx * TLEN + t) * 8);
    const int4 xa = xp[0];
    const int4 xb = xp[1];
    const unsigned* w12 = ws + OFF12D;
    const unsigned* w56 = ws + OFF56D;
    const uint2* w3 = (const uint2*)(ws + OFF3D);
    const uint2* w4 = (const uint2*)(ws + OFF4D);
    const uint2* w7 = (const uint2*)(ws + OFF7D);
    const uint2* w8 = (const uint2*)(ws + OFF8D);
    EmbP r;
    r.u0 = w12[xa.x * 2 + xa.y];
    const uint2 q3 = w3[xa.z];
    const uint2 q4 = w4[xa.w];
    r.u4 = w56[xb.x * 2 + xb.y];
    const uint2 q7 = w7[xb.z];
    const uint2 q8 = w8[xb.w];
    r.u1 = q3.x;                                   // e3_0, e3_1
    r.u2 = (q3.y & 0xffffu) | (q4.x << 16);        // e3_2, e4_0
    r.u3 = (q4.x >> 16)     | (q4.y << 16);        // e4_1, e4_2  (v_alignbit)
    r.u5 = q7.x;                                   // e7_0, e7_1
    r.u6 = (q7.y & 0xffffu) | (q8.x << 16);        // e7_2, e8_0
    r.u7 = (q8.x >> 16)     | (q8.y << 16);        // e8_1, e8_2
    return r;
}

// ---- One-block prep kernel: convert + fuse + pad embedding tables to f16 ----
__global__ void emb_pack_kernel(
    const float* __restrict__ e1, const float* __restrict__ e2,
    const float* __restrict__ e3, const float* __restrict__ e4,
    const float* __restrict__ e5, const float* __restrict__ e6,
    const float* __restrict__ e7, const float* __restrict__ e8,
    unsigned* __restrict__ ws)
{
    const int tid = threadIdx.x;
    if (tid < 4)  ws[OFF12D + tid] = pk2(e1[tid >> 1], e2[tid & 1]);
    if (tid < 10) ws[OFF56D + tid] = pk2(e5[tid / 2], e6[tid % 2]);
    for (int j = tid; j < 21; j += 128) {
        ws[OFF3D + 2 * j]     = pk2(e3[3 * j], e3[3 * j + 1]);
        ws[OFF3D + 2 * j + 1] = pk2(e3[3 * j + 2], 0.f);
    }
    for (int j = tid; j < 22; j += 128) {
        ws[OFF4D + 2 * j]     = pk2(e4[3 * j], e4[3 * j + 1]);
        ws[OFF4D + 2 * j + 1] = pk2(e4[3 * j + 2], 0.f);
    }
    for (int j = tid; j < 22; j += 128) {
        ws[OFF7D + 2 * j]     = pk2(e7[3 * j], e7[3 * j + 1]);
        ws[OFF7D + 2 * j + 1] = pk2(e7[3 * j + 2], 0.f);
    }
    for (int j = tid; j < 24; j += 128) {
        ws[OFF8D + 2 * j]     = pk2(e8[3 * j], e8[3 * j + 1]);
        ws[OFF8D + 2 * j + 1] = pk2(e8[3 * j + 2], 0.f);
    }
}

// One wave = 16 batch rows for all 64 timesteps.
// WMMA f32_16x16x32_f16: A = fused [Wih|Whh|pad] (M=16 permuted gate rows, K=32),
// B = [features ; h ; pad] (K=32, N=16 batch), D = gates (f32).
__global__ __launch_bounds__(128) void lstm2_wmma_kernel(
    const int*   __restrict__ x,
    const unsigned* __restrict__ ws,
    const float* __restrict__ Wih0, const float* __restrict__ Whh0,
    const float* __restrict__ bih0, const float* __restrict__ bhh0,
    const float* __restrict__ Wih1, const float* __restrict__ Whh1,
    const float* __restrict__ bih1, const float* __restrict__ bhh1,
    const float* __restrict__ fc6w, const float* __restrict__ fc6b,
    const float* __restrict__ fc7w, const float* __restrict__ fc7b,
    const float* __restrict__ fc1w, const float* __restrict__ fc1b,
    const float* __restrict__ fc2w, const float* __restrict__ fc2b,
    const float* __restrict__ fc3w, const float* __restrict__ fc3b,
    float* __restrict__ out)
{
    const int  lane   = threadIdx.x & 31;
    const int  gwave  = (int)((blockIdx.x * blockDim.x + threadIdx.x) >> 5);
    const int  m      = lane & 15;          // A: row M ; B/D: batch column N
    const bool hiHalf = (lane & 16) != 0;   // selects K-range (A/B) / M+8 (D)
    const bool loSel  = !hiHalf;
    const int  bidx   = gwave * 16 + m;     // this lane's batch element

    // ---- Gather A operands (weights -> f16, WMMA A layout), loop-invariant ----
    // 16-bit A 16x32 layout: lane m = M; element c : K = (hi?16:0)+c
    //                                    (contiguous K order, matches B)
    const int gIF = gate_of(m);   // row m of the i|f matrices
    const int gGO = 16 + gIF;     // row m of the g|o matrices
    v16h a0a, a0b, a1a, a1b;
#pragma unroll
    for (int c = 0; c < 16; ++c) {
        const int v = c >> 1;
        const int k = (hiHalf ? 16 : 0) + 2 * v + (c & 1);
        float w;
        // layer 0: K = [emb(16) | h1(8) | pad(8)]
        w = (k < 16) ? Wih0[gIF * 16 + k] : ((k < 24) ? Whh0[gIF * 8 + (k - 16)] : 0.f);
        a0a[c] = (_Float16)w;
        w = (k < 16) ? Wih0[gGO * 16 + k] : ((k < 24) ? Whh0[gGO * 8 + (k - 16)] : 0.f);
        a0b[c] = (_Float16)w;
        // layer 1: K = [h1(8) | h2(8) | pad(16)]
        w = (k < 8) ? Wih1[gIF * 8 + k] : ((k < 16) ? Whh1[gIF * 8 + (k - 8)] : 0.f);
        a1a[c] = (_Float16)w;
        w = (k < 8) ? Wih1[gGO * 8 + k] : ((k < 16) ? Whh1[gGO * 8 + (k - 8)] : 0.f);
        a1b[c] = (_Float16)w;
    }

    // ---- Bias accumulators in D layout: VGPR r -> row (r + (hi?8:0)) ----
    v8f cb0_if, cb0_go, cb1_if, cb1_go;
#pragma unroll
    for (int r = 0; r < 8; ++r) {
        const int gb = gate_of(r + (hiHalf ? 8 : 0));
        cb0_if[r] = bih0[gb]      + bhh0[gb];
        cb0_go[r] = bih0[16 + gb] + bhh0[16 + gb];
        cb1_if[r] = bih1[gb]      + bhh1[gb];
        cb1_go[r] = bih1[16 + gb] + bhh1[16 + gb];
    }

    // ---- Collapse fc6->fc7 and fc1->fc2 (purely linear) into 8-vectors ----
    float wth[8], whv[8];
#pragma unroll
    for (int j = 0; j < 8; ++j) {
        float s1 = 0.f, s2 = 0.f;
#pragma unroll
        for (int q = 0; q < 4; ++q) {
            s1 += fc7w[q] * fc6w[q * 8 + j];
            s2 += fc2w[q] * fc1w[q * 8 + j];
        }
        wth[j] = s1;
        whv[j] = s2;
    }
    float bth = fc7b[0], bhv = fc2b[0];
#pragma unroll
    for (int q = 0; q < 4; ++q) {
        bth += fc7w[q] * fc6b[q];
        bhv += fc2w[q] * fc1b[q];
    }

    // ---- Recurrent state ----
    // c-state: this lane's 4 hidden units (j0..j0+3, j0 = hi?4:0), f32.
    // h-state: full 8 units as 4 packed f16 pairs (K order), identical in
    //          both lane halves after the xor-16 exchange.
    float c1[4], c2[4];
    unsigned h1pk[4], h2pk[4];
#pragma unroll
    for (int q = 0; q < 4; ++q) { c1[q] = 0.f; c2[q] = 0.f; h1pk[q] = 0u; h2pk[q] = 0u; }
    float acc = 0.f;

    // Software pipeline: embedding pairs for step t are loaded during step t-1.
    EmbP er = load_emb(x, bidx, 0, ws);

    for (int t = 0; t < TLEN; ++t) {
        // ---- Layer-0 B operand (branchless): low lanes K=0..15 = features,
        //      high lanes K=16..23 = h1, K=24..31 = pad ----
        BOp b0;
        b0.u[0] = loSel ? er.u0 : h1pk[0];
        b0.u[1] = loSel ? er.u1 : h1pk[1];
        b0.u[2] = loSel ? er.u2 : h1pk[2];
        b0.u[3] = loSel ? er.u3 : h1pk[3];
        b0.u[4] = loSel ? er.u4 : 0u;
        b0.u[5] = loSel ? er.u5 : 0u;
        b0.u[6] = loSel ? er.u6 : 0u;
        b0.u[7] = loSel ? er.u7 : 0u;

        const v8f d_if = __builtin_amdgcn_wmma_f32_16x16x32_f16(
            false, a0a, false, b0.h, (short)0, cb0_if, false, false);
        const v8f d_go = __builtin_amdgcn_wmma_f32_16x16x32_f16(
            false, a0b, false, b0.h, (short)0, cb0_go, false, false);

        // Prefetch next step's embedding pairs (overlaps the update chain).
        er = load_emb(x, bidx, (t < TLEN - 1) ? t + 1 : t, ws);

        // Permuted rows: D[q]=i_{j0+q}, D[4+q]=f_{j0+q}; same for g|o matrix.
        float h1n[4];
#pragma unroll
        for (int q = 0; q < 4; ++q) {
            const float cn = fast_sig(d_if[4 + q]) * c1[q]
                           + fast_sig(d_if[q]) * fast_tanh(d_go[q]);
            c1[q] = cn;
            h1n[q] = fast_sig(d_go[4 + q]) * fast_tanh(cn);
        }

        // ---- Exchange h1 halves (2 packed shuffles -> full h1 everywhere) ----
        {
            const unsigned a01 = pk2(h1n[0], h1n[1]);
            const unsigned a23 = pk2(h1n[2], h1n[3]);
            const unsigned p01 = (unsigned)__shfl_xor((int)a01, 16, 32);
            const unsigned p23 = (unsigned)__shfl_xor((int)a23, 16, 32);
            h1pk[0] = hiHalf ? p01 : a01;
            h1pk[1] = hiHalf ? p23 : a23;
            h1pk[2] = hiHalf ? a01 : p01;
            h1pk[3] = hiHalf ? a23 : p23;
        }

        // ---- Layer-1 B: low lanes K=0..7 h1, K=8..15 h2(prev); high lanes pad ----
        BOp b1;
        b1.u[0] = loSel ? h1pk[0] : 0u;
        b1.u[1] = loSel ? h1pk[1] : 0u;
        b1.u[2] = loSel ? h1pk[2] : 0u;
        b1.u[3] = loSel ? h1pk[3] : 0u;
        b1.u[4] = loSel ? h2pk[0] : 0u;
        b1.u[5] = loSel ? h2pk[1] : 0u;
        b1.u[6] = loSel ? h2pk[2] : 0u;
        b1.u[7] = loSel ? h2pk[3] : 0u;

        const v8f e_if = __builtin_amdgcn_wmma_f32_16x16x32_f16(
            false, a1a, false, b1.h, (short)0, cb1_if, false, false);
        const v8f e_go = __builtin_amdgcn_wmma_f32_16x16x32_f16(
            false, a1b, false, b1.h, (short)0, cb1_go, false, false);

        float h2n[4];
#pragma unroll
        for (int q = 0; q < 4; ++q) {
            const float cn = fast_sig(e_if[4 + q]) * c2[q]
                           + fast_sig(e_if[q]) * fast_tanh(e_go[q]);
            c2[q] = cn;
            h2n[q] = fast_sig(e_go[4 + q]) * fast_tanh(cn);
        }

        // ---- Exchange h2 halves ----
        {
            const unsigned a01 = pk2(h2n[0], h2n[1]);
            const unsigned a23 = pk2(h2n[2], h2n[3]);
            const unsigned p01 = (unsigned)__shfl_xor((int)a01, 16, 32);
            const unsigned p23 = (unsigned)__shfl_xor((int)a23, 16, 32);
            h2pk[0] = hiHalf ? p01 : a01;
            h2pk[1] = hiHalf ? p23 : a23;
            h2pk[2] = hiHalf ? a01 : p01;
            h2pk[3] = hiHalf ? a23 : p23;
        }

        // ---- Collapsed FC heads + weighted time reduction ----
        float th = bth, hv = bhv;
#pragma unroll
        for (int p = 0; p < 4; ++p) {
            const float lo = unpk0(h2pk[p]);
            const float hi = unpk1(h2pk[p]);
            th += wth[2 * p] * lo + wth[2 * p + 1] * hi;
            hv += whv[2 * p] * lo + whv[2 * p + 1] * hi;
        }
        acc += fc3w[t] * th * hv;
    }

    if (lane < 16) out[bidx] = acc + fc3b[0];
}

extern "C" void kernel_launch(void* const* d_in, const int* in_sizes, int n_in,
                              void* d_out, int out_size, void* d_ws, size_t ws_size,
                              hipStream_t stream) {
    (void)in_sizes; (void)n_in; (void)out_size; (void)ws_size;
    const int*   x    = (const int*)  d_in[0];
    const float* e1   = (const float*)d_in[1];
    const float* e2   = (const float*)d_in[2];
    const float* e3   = (const float*)d_in[3];
    const float* e4   = (const float*)d_in[4];
    const float* e5   = (const float*)d_in[5];
    const float* e6   = (const float*)d_in[6];
    const float* e7   = (const float*)d_in[7];
    const float* e8   = (const float*)d_in[8];
    const float* Wih0 = (const float*)d_in[9];
    const float* Whh0 = (const float*)d_in[10];
    const float* bih0 = (const float*)d_in[11];
    const float* bhh0 = (const float*)d_in[12];
    const float* Wih1 = (const float*)d_in[13];
    const float* Whh1 = (const float*)d_in[14];
    const float* bih1 = (const float*)d_in[15];
    const float* bhh1 = (const float*)d_in[16];
    const float* fc6w = (const float*)d_in[17];
    const float* fc6b = (const float*)d_in[18];
    const float* fc7w = (const float*)d_in[19];
    const float* fc7b = (const float*)d_in[20];
    const float* fc1w = (const float*)d_in[21];
    const float* fc1b = (const float*)d_in[22];
    const float* fc2w = (const float*)d_in[23];
    const float* fc2b = (const float*)d_in[24];
    const float* fc3w = (const float*)d_in[25];
    const float* fc3b = (const float*)d_in[26];
    unsigned* ws = (unsigned*)d_ws;
    float* out = (float*)d_out;

    // Pack embedding tables to fused/padded f16 in workspace (<1 KB).
    emb_pack_kernel<<<1, 128, 0, stream>>>(e1, e2, e3, e4, e5, e6, e7, e8, ws);

    // 16384 batch / 16 per wave = 1024 waves; 128-thread blocks = 4 waves/block.
    const int nblocks = (BATCH / 16) / 4;
    lstm2_wmma_kernel<<<nblocks, 128, 0, stream>>>(
        x, ws,
        Wih0, Whh0, bih0, bhh0, Wih1, Whh1, bih1, bhh1,
        fc6w, fc6b, fc7w, fc7b, fc1w, fc1b, fc2w, fc2b, fc3w, fc3b,
        out);
}
